// TemporalGNNCell_38989713113511
// MI455X (gfx1250) — matrix operand order repaired and checked by reading, not compiled, not run on hardware
//
#include <hip/hip_runtime.h>
#include <math.h>

#define DEV_INLINE __device__ __forceinline__

typedef __attribute__((ext_vector_type(16))) _Float16 v16h;
typedef __attribute__((ext_vector_type(8)))  _Float16 v8h;
typedef __attribute__((ext_vector_type(8)))  float    v8f;

constexpr int  B_ = 4, N_ = 10000, E_ = 160000, F_ = 64, H_ = 128, HEADS_ = 4, D_ = 32, ED_ = 16;
constexpr int  BN_ = B_ * N_;
constexpr float NEG_SLOPE = 0.2f;
constexpr float LN_EPS = 1e-5f;

// ---------------------------------------------------------------------------
// WMMA fragment loaders from pre-converted f16 operands.
// A-frag (16x32, MxK): lane half hi covers K = {k0+8hi .. +7} and {k0+16+8hi .. +7}
//   -> two aligned 16B loads.
// B-frag (32x16, KxN) from W rows (B[k][n] = W[n][k]): lane half hi covers
//   K = {k0+8hi .. +7} and {k0+16+8hi .. +7} -> two aligned 16B loads.
// ---------------------------------------------------------------------------
DEV_INLINE v16h load_a_frag(const _Float16* __restrict__ row, int k0, int hi) {
  v8h lo = *reinterpret_cast<const v8h*>(row + k0 + 8 * hi);
  v8h hh = *reinterpret_cast<const v8h*>(row + k0 + 16 + 8 * hi);
  v16h f;
#pragma unroll
  for (int j = 0; j < 8; ++j) { f[j] = lo[j]; f[8 + j] = hh[j]; }
  return f;
}

DEV_INLINE v16h load_b_frag(const _Float16* __restrict__ row, int k0, int hi) {
  v8h lo = *reinterpret_cast<const v8h*>(row + k0 + 8 * hi);
  v8h hh = *reinterpret_cast<const v8h*>(row + k0 + 16 + 8 * hi);
  v16h f;
#pragma unroll
  for (int j = 0; j < 8; ++j) { f[j] = lo[j]; f[8 + j] = hh[j]; }
  return f;
}

// out[M,NN] = X1[M,K1] @ W1[NN,K1]^T (+ X2[M,K2] @ W2[NN,K2]^T) (+ bias1 + bias2)
// Register blocking: each wave owns 2 M-tiles x NT N-tiles (2*NT independent
// accumulators -> WMMA->WMMA RAW hazard hidden, ~1 fragment load per WMMA).
// 8 waves / block. grid = (M/32, NN/(128*NT)). All dims exact multiples.
template <int NT>
__global__ __launch_bounds__(256) void k_gemm_xwT(
    const _Float16* __restrict__ X1, const _Float16* __restrict__ W1, int K1,
    const _Float16* __restrict__ X2, const _Float16* __restrict__ W2, int K2,
    const float* __restrict__ bias1, const float* __restrict__ bias2,
    float* __restrict__ out, int NN) {
  const int lane = threadIdx.x & 31;
  const int wave = threadIdx.x >> 5;
  const int m0 = blockIdx.x * 32;                    // two m-tiles: m0, m0+16
  const int n0 = (blockIdx.y * 8 + wave) * (16 * NT);
  const int r  = lane & 15;
  const int hi = lane >> 4;

  v8f acc[2][NT];
#pragma unroll
  for (int i = 0; i < 2; ++i)
#pragma unroll
    for (int j = 0; j < NT; ++j) acc[i][j] = (v8f){0.f, 0.f, 0.f, 0.f, 0.f, 0.f, 0.f, 0.f};

  {
    const _Float16* arow0 = X1 + (size_t)(m0 + r) * K1;
    const _Float16* arow1 = X1 + (size_t)(m0 + 16 + r) * K1;
    for (int k0 = 0; k0 < K1; k0 += 32) {
      __builtin_prefetch(arow0 + k0 + 64, 0, 1);     // global_prefetch_b8: stream A rows
      v16h a0 = load_a_frag(arow0, k0, hi);
      v16h a1 = load_a_frag(arow1, k0, hi);
#pragma unroll
      for (int j = 0; j < NT; ++j) {
        v16h b = load_b_frag(W1 + (size_t)(n0 + 16 * j + r) * K1, k0, hi);
        acc[0][j] = __builtin_amdgcn_wmma_f32_16x16x32_f16(false, a0, false, b, (short)0,
                                                           acc[0][j], false, false);
        acc[1][j] = __builtin_amdgcn_wmma_f32_16x16x32_f16(false, a1, false, b, (short)0,
                                                           acc[1][j], false, false);
      }
    }
  }
  if (X2 != nullptr) {
    const _Float16* arow0 = X2 + (size_t)(m0 + r) * K2;
    const _Float16* arow1 = X2 + (size_t)(m0 + 16 + r) * K2;
    for (int k0 = 0; k0 < K2; k0 += 32) {
      v16h a0 = load_a_frag(arow0, k0, hi);
      v16h a1 = load_a_frag(arow1, k0, hi);
#pragma unroll
      for (int j = 0; j < NT; ++j) {
        v16h b = load_b_frag(W2 + (size_t)(n0 + 16 * j + r) * K2, k0, hi);
        acc[0][j] = __builtin_amdgcn_wmma_f32_16x16x32_f16(false, a0, false, b, (short)0,
                                                           acc[0][j], false, false);
        acc[1][j] = __builtin_amdgcn_wmma_f32_16x16x32_f16(false, a1, false, b, (short)0,
                                                           acc[1][j], false, false);
      }
    }
  }

#pragma unroll
  for (int j = 0; j < NT; ++j) {
    const int n = n0 + 16 * j + r;
    float bv = 0.f;
    if (bias1) bv += bias1[n];
    if (bias2) bv += bias2[n];
#pragma unroll
    for (int mt = 0; mt < 2; ++mt)
#pragma unroll
      for (int i = 0; i < 8; ++i)     // C/D layout: m = vgpr + 8*hi, n = lane&15
        out[(size_t)(m0 + 16 * mt + i + 8 * hi) * NN + n] = acc[mt][j][i] + bv;
  }
}

// ---------------------------------------------------------------------------
// Small prep / elementwise kernels
// ---------------------------------------------------------------------------
__global__ void k_fill(float* __restrict__ p, float v, int n) {
  int t = blockIdx.x * blockDim.x + threadIdx.x;
  if (t < n) p[t] = v;
}

// generic fp32 -> f16 converter
__global__ void k_cvt(const float* __restrict__ in, _Float16* __restrict__ out, int n) {
  int t = blockIdx.x * blockDim.x + threadIdx.x;
  if (t < n) out[t] = (_Float16)in[t];
}

// W_lin [F,H] -> WlinT [H,F] in f16
__global__ void k_transpose_wlin(const float* __restrict__ W, _Float16* __restrict__ WT) {
  int t = blockIdx.x * blockDim.x + threadIdx.x;   // F*H = 8192
  if (t >= F_ * H_) return;
  int i = t >> 7, j = t & (H_ - 1);
  WT[j * F_ + i] = (_Float16)W[i * H_ + j];
}

// wcomb[k,hh] = sum_d W_edge[k, hh*D+d] * att_edge[hh,d]   (16x4)
__global__ void k_wcomb(const float* __restrict__ We, const float* __restrict__ ae,
                        float* __restrict__ wc) {
  int t = threadIdx.x;                              // 64 threads
  if (t >= ED_ * HEADS_) return;
  int k = t >> 2, hh = t & 3;
  float s = 0.f;
  for (int d = 0; d < D_; ++d) s += We[k * H_ + hh * D_ + d] * ae[hh * D_ + d];
  wc[k * HEADS_ + hh] = s;
}

// a_src/a_dst per (b,n,head)
__global__ void k_node_att(const float* __restrict__ h, const float* __restrict__ as,
                           const float* __restrict__ ad, float* __restrict__ o_s,
                           float* __restrict__ o_d) {
  int t = blockIdx.x * blockDim.x + threadIdx.x;   // BN*HEADS
  if (t >= BN_ * HEADS_) return;
  int bn = t >> 2, hh = t & 3;
  const float* hp = h + (size_t)bn * H_ + hh * D_;
  float s1 = 0.f, s2 = 0.f;
  for (int d = 0; d < D_; ++d) { float v = hp[d]; s1 += v * as[hh * D_ + d]; s2 += v * ad[hh * D_ + d]; }
  o_s[t] = s1; o_d[t] = s2;
}

// a_edge[e,hh] = edge_attr[e,:] . wcomb[:,hh]
__global__ void k_a_edge(const float* __restrict__ ea, const float* __restrict__ wc,
                         float* __restrict__ out) {
  int t = blockIdx.x * blockDim.x + threadIdx.x;   // E*HEADS
  if (t >= E_ * HEADS_) return;
  int e = t >> 2, hh = t & 3;
  const float* row = ea + (size_t)e * ED_;
  float s = 0.f;
  for (int k = 0; k < ED_; ++k) s += row[k] * wc[k * HEADS_ + hh];
  out[t] = s;
}

DEV_INLINE void atomicMaxFloat(float* addr, float val) {
  if (val >= 0.f) atomicMax(reinterpret_cast<int*>(addr), __float_as_int(val));
  else            atomicMin(reinterpret_cast<unsigned int*>(addr), __float_as_uint(val));
}

__global__ void k_edge_logits(const int* __restrict__ src, const int* __restrict__ dst,
                              const unsigned char* __restrict__ mask,
                              const float* __restrict__ a_s, const float* __restrict__ a_d,
                              const float* __restrict__ a_e, float* __restrict__ lg,
                              float* __restrict__ maxb) {
  int t = blockIdx.x * blockDim.x + threadIdx.x;   // E*16
  if (t >= E_ * 16) return;
  int e = t >> 4, c = t & 15, b = c >> 2, hh = c & 3;
  int s = src[e], d = dst[e];
  float l = a_s[(b * N_ + s) * HEADS_ + hh] + a_d[(b * N_ + d) * HEADS_ + hh] +
            a_e[e * HEADS_ + hh];
  l = (l > 0.f) ? l : NEG_SLOPE * l;
  if (!mask[e]) l = -1e9f;
  lg[t] = l;
  atomicMaxFloat(&maxb[d * 16 + c], l);
}

__global__ void k_edge_exp(const int* __restrict__ dst, float* __restrict__ lg,
                           const float* __restrict__ maxb, float* __restrict__ den) {
  int t = blockIdx.x * blockDim.x + threadIdx.x;   // E*16
  if (t >= E_ * 16) return;
  int e = t >> 4, c = t & 15;
  int d = dst[e];
  float v = expf(lg[t] - maxb[d * 16 + c]);
  lg[t] = v;                                       // in-place: lg now holds exp
  atomicAdd(&den[d * 16 + c], v);
}

// weighted scatter: agg[dst, b, hh*D+dch] += alpha * h[b, src, hh*D+dch]
__global__ void k_aggregate(const int* __restrict__ src, const int* __restrict__ dst,
                            const float* __restrict__ ex, const float* __restrict__ den,
                            const float* __restrict__ h, float* __restrict__ agg) {
  int t = blockIdx.x * blockDim.x + threadIdx.x;   // E*512; one wave = one (e, b, head)
  if (t >= E_ * 512) return;
  int e = t >> 9, rr = t & 511, c = rr >> 5, dch = rr & 31;
  int b = c >> 2, hh = c & 3;
  int dn = dst[e], sn = src[e];
  float alpha = ex[e * 16 + c] / (den[dn * 16 + c] + 1e-16f);
  float v = alpha * h[((size_t)b * N_ + sn) * H_ + hh * D_ + dch];
  atomicAdd(&agg[((size_t)dn * B_ + b) * H_ + hh * D_ + dch], v);
}

// X[b*N+n, j] = agg[n, b, j] + gat_bias[j]   (written directly in f16 for WMMA)
__global__ void k_build_X(const float* __restrict__ agg, const float* __restrict__ bias,
                          _Float16* __restrict__ X) {
  int t = blockIdx.x * blockDim.x + threadIdx.x;   // BN*H
  if (t >= BN_ * H_) return;
  int j = t & (H_ - 1), bn = t >> 7;
  int b = bn / N_, n = bn - b * N_;
  X[t] = (_Float16)(agg[((size_t)n * B_ + b) * H_ + j] + bias[j]);
}

DEV_INLINE float sigf(float x) { return 1.f / (1.f + expf(-x)); }

// LSTM gate activations; optionally emits an f16 copy of h_new (next GEMM's A).
__global__ void k_lstm_act(const float* __restrict__ gates, const float* __restrict__ c0,
                           float* __restrict__ h_new, float* __restrict__ c_new,
                           _Float16* __restrict__ h_half) {
  int t = blockIdx.x * blockDim.x + threadIdx.x;   // BN*H
  if (t >= BN_ * H_) return;
  int m = t >> 7, j = t & (H_ - 1);
  const float* g = gates + (size_t)m * (4 * H_);
  float i = sigf(g[j]);
  float f = sigf(g[H_ + j]);
  float gg = tanhf(g[2 * H_ + j]);
  float o = sigf(g[3 * H_ + j]);
  float c = f * c0[t] + i * gg;
  c_new[t] = c;
  float hv = o * tanhf(c);
  h_new[t] = hv;
  if (h_half) h_half[t] = (_Float16)hv;
}

// one wave per row of 128; wave32 shuffle reduction
__global__ __launch_bounds__(256) void k_layernorm(const float* __restrict__ h2,
                                                   const float* __restrict__ gamma,
                                                   const float* __restrict__ beta,
                                                   float* __restrict__ out) {
  int wave = threadIdx.x >> 5, lane = threadIdx.x & 31;
  int m = blockIdx.x * 8 + wave;                   // BN rows, grid = BN/8
  const float4* xp = reinterpret_cast<const float4*>(h2 + (size_t)m * H_) + lane;
  float4 x = *xp;
  float s = x.x + x.y + x.z + x.w;
  float ss = x.x * x.x + x.y * x.y + x.z * x.z + x.w * x.w;
#pragma unroll
  for (int off = 16; off > 0; off >>= 1) {
    s  += __shfl_xor(s, off);
    ss += __shfl_xor(ss, off);
  }
  float mean = s * (1.f / H_);
  float var = ss * (1.f / H_) - mean * mean;
  float inv = rsqrtf(var + LN_EPS);
  float4 gm = reinterpret_cast<const float4*>(gamma)[lane];
  float4 bt = reinterpret_cast<const float4*>(beta)[lane];
  float4 o;
  o.x = (x.x - mean) * inv * gm.x + bt.x;
  o.y = (x.y - mean) * inv * gm.y + bt.y;
  o.z = (x.z - mean) * inv * gm.z + bt.z;
  o.w = (x.w - mean) * inv * gm.w + bt.w;
  reinterpret_cast<float4*>(out + (size_t)m * H_)[lane] = o;
}

// ---------------------------------------------------------------------------
extern "C" void kernel_launch(void* const* d_in, const int* in_sizes, int n_in,
                              void* d_out, int out_size, void* d_ws, size_t ws_size,
                              hipStream_t stream) {
  const float* x        = (const float*)d_in[0];
  const int*   eidx     = (const int*)d_in[1];
  const float* eattr    = (const float*)d_in[2];
  const unsigned char* emask = (const unsigned char*)d_in[3];
  const float* h0       = (const float*)d_in[4];
  const float* c0       = (const float*)d_in[5];
  const float* Wlin     = (const float*)d_in[6];
  const float* att_src  = (const float*)d_in[7];
  const float* att_dst  = (const float*)d_in[8];
  const float* Wedge    = (const float*)d_in[9];
  const float* att_edge = (const float*)d_in[10];
  const float* gat_bias = (const float*)d_in[11];
  const float* Wih0     = (const float*)d_in[12];
  const float* Whh0     = (const float*)d_in[13];
  const float* bih0     = (const float*)d_in[14];
  const float* bhh0     = (const float*)d_in[15];
  const float* Wih1     = (const float*)d_in[16];
  const float* Whh1     = (const float*)d_in[17];
  const float* bih1     = (const float*)d_in[18];
  const float* bhh1     = (const float*)d_in[19];
  const float* gamma    = (const float*)d_in[20];
  const float* beta     = (const float*)d_in[21];

  const int* src = eidx;
  const int* dst = eidx + E_;

  float* out = (float*)d_out;
  float* h_out = out;                    // [B,N,H]
  float* h1 = out + (size_t)BN_ * H_;    // h_new[0]
  float* h2 = h1 + (size_t)BN_ * H_;     // h_new[1]
  float* c1 = h2 + (size_t)BN_ * H_;     // c_new[0]
  float* c2 = c1 + (size_t)BN_ * H_;     // c_new[1]

  // ---------------- workspace layout ----------------
  // fp32 region first, then f16 region (all sections 16B-aligned).
  float* ws = (float*)d_ws;
  float* wcomb  = ws;                          // 64
  float* h_ws   = wcomb + 64;                  // BN*H
  float* a_s    = h_ws + (size_t)BN_ * H_;     // BN*HEADS
  float* a_d    = a_s + BN_ * HEADS_;          // BN*HEADS
  float* a_e    = a_d + BN_ * HEADS_;          // E*HEADS
  float* lg     = a_e + E_ * HEADS_;           // E*16 (logits, then exp in place)
  float* maxb   = lg + (size_t)E_ * 16;        // N*16
  float* den    = maxb + N_ * 16;              // N*16
  float* agg    = den + N_ * 16;               // N*B*H
  float* gates  = agg + (size_t)N_ * B_ * H_;  // BN*4H (reused by both layers)

  _Float16* x_h     = (_Float16*)(gates + (size_t)BN_ * 4 * H_);  // BN*F
  _Float16* WlinT_h = x_h + (size_t)BN_ * F_;                     // H*F
  _Float16* h0_h    = WlinT_h + H_ * F_;                          // 2*BN*H
  _Float16* Wih0_h  = h0_h + (size_t)2 * BN_ * H_;                // 4H*H each
  _Float16* Whh0_h  = Wih0_h + 4 * H_ * H_;
  _Float16* Wih1_h  = Whh0_h + 4 * H_ * H_;
  _Float16* Whh1_h  = Wih1_h + 4 * H_ * H_;
  _Float16* Xbuf_h  = Whh1_h + 4 * H_ * H_;                       // BN*H
  _Float16* h1_h    = Xbuf_h + (size_t)BN_ * H_;                  // BN*H

  const int T = 256;
  // --- f16 operand prep (weights tiny, streams fully parallel) ---
  k_transpose_wlin<<<(F_ * H_ + T - 1) / T, T, 0, stream>>>(Wlin, WlinT_h);
  k_cvt<<<(BN_ * F_ + T - 1) / T, T, 0, stream>>>(x, x_h, BN_ * F_);
  k_cvt<<<(2 * BN_ * H_ + T - 1) / T, T, 0, stream>>>(h0, h0_h, 2 * BN_ * H_);
  k_cvt<<<(4 * H_ * H_ + T - 1) / T, T, 0, stream>>>(Wih0, Wih0_h, 4 * H_ * H_);
  k_cvt<<<(4 * H_ * H_ + T - 1) / T, T, 0, stream>>>(Whh0, Whh0_h, 4 * H_ * H_);
  k_cvt<<<(4 * H_ * H_ + T - 1) / T, T, 0, stream>>>(Wih1, Wih1_h, 4 * H_ * H_);
  k_cvt<<<(4 * H_ * H_ + T - 1) / T, T, 0, stream>>>(Whh1, Whh1_h, 4 * H_ * H_);
  k_wcomb<<<1, 64, 0, stream>>>(Wedge, att_edge, wcomb);

  // --- GAT ---
  // h = x @ W_lin  (M=40000, K=64, NN=128) : NT=1 (block covers 128 cols)
  k_gemm_xwT<1><<<dim3(BN_ / 32, 1), T, 0, stream>>>(
      x_h, WlinT_h, F_, nullptr, nullptr, 0, nullptr, nullptr, h_ws, H_);
  k_node_att<<<(BN_ * HEADS_ + T - 1) / T, T, 0, stream>>>(h_ws, att_src, att_dst, a_s, a_d);
  k_a_edge<<<(E_ * HEADS_ + T - 1) / T, T, 0, stream>>>(eattr, wcomb, a_e);
  k_fill<<<(N_ * 16 + T - 1) / T, T, 0, stream>>>(maxb, -INFINITY, N_ * 16);
  k_fill<<<(N_ * 16 + T - 1) / T, T, 0, stream>>>(den, 0.f, N_ * 16);
  k_fill<<<(N_ * B_ * H_ + T - 1) / T, T, 0, stream>>>(agg, 0.f, N_ * B_ * H_);
  k_edge_logits<<<(E_ * 16 + T - 1) / T, T, 0, stream>>>(src, dst, emask, a_s, a_d, a_e, lg, maxb);
  k_edge_exp<<<(E_ * 16 + T - 1) / T, T, 0, stream>>>(dst, lg, maxb, den);
  k_aggregate<<<(E_ * 512 + T - 1) / T, T, 0, stream>>>(src, dst, lg, den, h_ws, agg);
  k_build_X<<<(BN_ * H_ + T - 1) / T, T, 0, stream>>>(agg, gat_bias, Xbuf_h);

  // --- LSTM layer 0: gates = X@Wih0^T + h0[0]@Whh0^T + bih0 + bhh0 ---
  // NT=2: block covers 256 cols, grid.y = 512/256 = 2
  k_gemm_xwT<2><<<dim3(BN_ / 32, 2), T, 0, stream>>>(
      Xbuf_h, Wih0_h, H_, h0_h, Whh0_h, H_, bih0, bhh0, gates, 4 * H_);
  k_lstm_act<<<(BN_ * H_ + T - 1) / T, T, 0, stream>>>(gates, c0, h1, c1, h1_h);

  // --- LSTM layer 1 ---
  k_gemm_xwT<2><<<dim3(BN_ / 32, 2), T, 0, stream>>>(
      h1_h, Wih1_h, H_, h0_h + (size_t)BN_ * H_, Whh1_h, H_, bih1, bhh1, gates, 4 * H_);
  k_lstm_act<<<(BN_ * H_ + T - 1) / T, T, 0, stream>>>(gates, c0 + (size_t)BN_ * H_, h2, c2,
                                                       nullptr);

  // --- LayerNorm ---
  k_layernorm<<<BN_ / 8, T, 0, stream>>>(h2, gamma, beta, h_out);
}
